// Transformer_4612794875934
// MI455X (gfx1250) — matrix operand order repaired and checked by reading, not compiled, not run on hardware
//
#include <hip/hip_runtime.h>
#include <hip/hip_bf16.h>

// ---------------------------------------------------------------------------
// Transformer forward (encoder-decoder) for MI455X / gfx1250.
// f16 activation/weight pipeline: all GEMMs run v_wmma_f32_16x16x32_f16 with
// fp32 accumulation; A-tiles staged into LDS by the Tensor Data Mover
// (tensor_load_to_lds + s_wait_tensorcnt) using the D# pad feature to
// reproduce the padded LDS layout. wave32 fragment layouts per ISA 7.12.2.
// ---------------------------------------------------------------------------

#define DMODEL 512
#define NHEAD  8
#define DEPTH  64
#define FFDIM  2048
#define VOCAB  32000
#define BATCH  2
#define SEQ    1024
#define NLAYER 6
#define ROWS   (BATCH * SEQ)   // 2048

typedef _Float16 h8   __attribute__((ext_vector_type(8)));
typedef _Float16 v16h __attribute__((ext_vector_type(16)));
typedef float    v8f  __attribute__((ext_vector_type(8)));
typedef unsigned int u32x4 __attribute__((ext_vector_type(4)));
typedef int          i32x4 __attribute__((ext_vector_type(4)));
typedef int          i32x8 __attribute__((ext_vector_type(8)));

#define GF_BT     1   // B supplied as [N,K] (transposed view)
#define GF_RELU   2
#define GF_OUTF32 4   // write C as f32 (final logits)

#define LDSK 72       // 64 + 8 halves padding (144B row, keeps 16B alignment)

#if defined(__HIP_DEVICE_COMPILE__) && defined(__has_builtin)
#if __has_builtin(__builtin_amdgcn_tensor_load_to_lds)
#define HAVE_TDM 1
#endif
#endif

// ---------------------------------------------------------------------------
// TDM: DMA a 128-row x 64-half tile (row stride lda halves) into LDS at
// lds_off, inserting 4 DWORDs of padding after every 32 DWORDs (=> row pitch
// 72 halves, matching As[128][LDSK]).  D# per CDNA5 ISA ch.8.
// ---------------------------------------------------------------------------
__device__ __forceinline__ void tdm_load_tile_A(unsigned lds_off,
                                                const _Float16* gptr, int lda)
{
#ifdef HAVE_TDM
    const unsigned long long ga = (unsigned long long)gptr;
    const unsigned dim0 = 1u << 20, dim1 = 1u << 20;   // large => no OOB clip
    u32x4 g0 = {
        1u,                                            // count=1 (valid D#)
        lds_off,                                       // lds_addr
        (unsigned)ga,                                  // global_addr[31:0]
        (unsigned)((ga >> 32) & 0x01FFFFFFu) | (2u << 30)  // ga[56:32] | type=2
    };
    i32x8 g1 = {
        // data_size=1(2B) | pad_enable | pad_interval=4(32 dw) | pad_amount=3(4 dw)
        (int)((1u << 16) | (1u << 20) | (4u << 22) | (3u << 25)),
        (int)((dim0 & 0xFFFFu) << 16),                             // tensor_dim0 lo
        (int)(((dim0 >> 16) & 0xFFFFu) | ((dim1 & 0xFFFFu) << 16)),// dim0 hi|dim1 lo
        (int)(((dim1 >> 16) & 0xFFFFu) | (64u << 16)),             // dim1 hi|tile_dim0=64
        128,                                                       // tile_dim1=128 rows
        lda,                                                       // dim0_stride lo
        0, 0
    };
    i32x4 g2 = {0, 0, 0, 0};
    i32x4 g3 = {0, 0, 0, 0};
#if __clang_major__ >= 23
    i32x8 g4 = {0, 0, 0, 0, 0, 0, 0, 0};
    __builtin_amdgcn_tensor_load_to_lds(g0, g1, g2, g3, g4, 0);
#else
    __builtin_amdgcn_tensor_load_to_lds(g0, g1, g2, g3, 0);
#endif
#else
    (void)lds_off; (void)gptr; (void)lda;
#endif
}

// ---------------------------------------------------------------------------
// WMMA GEMM: C[b,h] = act(alpha * A[b,h] @ B[b,h] + bias)
// Block: 256 threads (8 waves). Tile 128(M) x 64(N) x 64(K-stage).
// Waves 4(M) x 2(N); each wave owns 32x32 of C -> 8 WMMA per stage.
// A tile staged by TDM (wave 0), B tile transposed into LDS by all waves.
// ---------------------------------------------------------------------------
__global__ __launch_bounds__(256) void wmma_gemm_kernel(
    const _Float16* __restrict__ A, const _Float16* __restrict__ Bm, void* __restrict__ Cv,
    int M, int N, int K, int lda, int ldb, int ldc,
    long long sAb, long long sAh, long long sBb, long long sBh,
    long long sCb, long long sCh, int nh,
    float alpha, const float* __restrict__ bias, int flags)
{
    __shared__ _Float16 As[128][LDSK];   // [m][k]
    __shared__ _Float16 BsT[64][LDSK];   // [n][k] (n-major: B frags contiguous)

    const int z  = blockIdx.z;
    const int bb = z / nh;
    const int hh = z - bb * nh;
    A  += bb * sAb + hh * sAh;
    Bm += bb * sBb + hh * sBh;
    const long long cOff = bb * sCb + hh * sCh;

    const int t    = threadIdx.x;
    const int lane = t & 31;
    const int wave = t >> 5;
    const int wm   = wave >> 1;     // 0..3  (M groups of 32)
    const int wn   = wave & 1;      // 0..1  (N groups of 32)
    const int m0   = blockIdx.y * 128;
    const int n0   = blockIdx.x * 64;

    const bool bt = (flags & GF_BT) != 0;

    // cooperative-load coordinates: chunk base = t>>3, in-row offset (t&7)*8
    const int crow = t >> 3;          // 0..31
    const int coff = (t & 7) * 8;     // 0,8,..,56

    const unsigned ldsA = (unsigned)(unsigned long long)&As[0][0];

    v8f c00 = {0.f,0.f,0.f,0.f,0.f,0.f,0.f,0.f};
    v8f c01 = c00, c10 = c00, c11 = c00;

    for (int kt = 0; kt < K; kt += 64) {
        __syncthreads();
        // ---- stage A 128x64 halves ----
#ifdef HAVE_TDM
        if (wave == 0)
            tdm_load_tile_A(ldsA, A + (long long)m0 * lda + kt, lda);
#else
        #pragma unroll
        for (int i = 0; i < 4; ++i) {
            const int r = crow + 32 * i;               // 0..127
            *(h8*)&As[r][coff] =
                *(const h8*)(A + (long long)(m0 + r) * lda + kt + coff);
        }
#endif
        if (kt + 64 < K)
            __builtin_prefetch(A + (long long)(m0 + crow) * lda + kt + 64 + coff, 0, 1);
        // ---- stage B 64(K)x64(N) halves as BsT[n][k] ----
        if (!bt) {
            #pragma unroll
            for (int i = 0; i < 2; ++i) {
                const int k = crow + 32 * i;           // 0..63
                h8 v = *(const h8*)(Bm + (long long)(kt + k) * ldb + n0 + coff);
                #pragma unroll
                for (int j = 0; j < 8; ++j) BsT[coff + j][k] = v[j];
            }
        } else {
            #pragma unroll
            for (int i = 0; i < 2; ++i) {
                const int n = crow + 32 * i;           // 0..63
                *(h8*)&BsT[n][coff] =
                    *(const h8*)(Bm + (long long)(n0 + n) * ldb + kt + coff);
            }
        }
#ifdef HAVE_TDM
        if (wave == 0)
            __builtin_amdgcn_s_wait_tensorcnt(0);
#endif
        __syncthreads();

        // ---- 2 x 32-K WMMA steps ----
        #pragma unroll
        for (int kk = 0; kk < 64; kk += 32) {
            // A 16x32 frag: lane m(0..15) row m K{0..7,16..23}; lane m+16 K{8..15,24..31}
            const int am  = wm * 32 + (lane & 15);
            const int akb = kk + ((lane & 16) ? 8 : 0);
            h8 a0lo = *(const h8*)&As[am][akb];
            h8 a0hi = *(const h8*)&As[am][akb + 16];
            h8 a1lo = *(const h8*)&As[am + 16][akb];
            h8 a1hi = *(const h8*)&As[am + 16][akb + 16];
            // B 32x16 frag: lane n(0..15) col n K{0..7,16..23}; lane n+16 K{8..15,24..31}
            const int bn  = wn * 32 + (lane & 15);
            const int bkb = kk + ((lane & 16) ? 8 : 0);
            h8 b0lo = *(const h8*)&BsT[bn][bkb];
            h8 b0hi = *(const h8*)&BsT[bn][bkb + 16];
            h8 b1lo = *(const h8*)&BsT[bn + 16][bkb];
            h8 b1hi = *(const h8*)&BsT[bn + 16][bkb + 16];

            v16h af0 = __builtin_shufflevector(a0lo, a0hi, 0,1,2,3,4,5,6,7,8,9,10,11,12,13,14,15);
            v16h af1 = __builtin_shufflevector(a1lo, a1hi, 0,1,2,3,4,5,6,7,8,9,10,11,12,13,14,15);
            v16h bf0 = __builtin_shufflevector(b0lo, b0hi, 0,1,2,3,4,5,6,7,8,9,10,11,12,13,14,15);
            v16h bf1 = __builtin_shufflevector(b1lo, b1hi, 0,1,2,3,4,5,6,7,8,9,10,11,12,13,14,15);

            c00 = __builtin_amdgcn_wmma_f32_16x16x32_f16(false, af0, false, bf0, (short)0, c00, false, false);
            c01 = __builtin_amdgcn_wmma_f32_16x16x32_f16(false, af0, false, bf1, (short)0, c01, false, false);
            c10 = __builtin_amdgcn_wmma_f32_16x16x32_f16(false, af1, false, bf0, (short)0, c10, false, false);
            c11 = __builtin_amdgcn_wmma_f32_16x16x32_f16(false, af1, false, bf1, (short)0, c11, false, false);
        }
    }

    // ---- epilogue: C/D layout — lane col = lane&15, rows (lane>=16?8:0)+r ----
    const int  col0  = n0 + wn * 32 + (lane & 15);
    const int  rbase = m0 + wm * 32 + ((lane & 16) ? 8 : 0);
    const bool relu  = (flags & GF_RELU) != 0;
    const float bv0  = bias ? bias[col0]      : 0.0f;
    const float bv1  = bias ? bias[col0 + 16] : 0.0f;
    #pragma unroll
    for (int r = 0; r < 8; ++r) {
        float v00 = alpha * c00[r] + bv0;
        float v01 = alpha * c01[r] + bv1;
        float v10 = alpha * c10[r] + bv0;
        float v11 = alpha * c11[r] + bv1;
        if (relu) {
            v00 = fmaxf(v00, 0.f); v01 = fmaxf(v01, 0.f);
            v10 = fmaxf(v10, 0.f); v11 = fmaxf(v11, 0.f);
        }
        const long long r0 = (long long)(rbase + r) * ldc;
        const long long r1 = (long long)(rbase + 16 + r) * ldc;
        if (flags & GF_OUTF32) {
            float* C = (float*)Cv + cOff;
            C[r0 + col0] = v00; C[r0 + col0 + 16] = v01;
            C[r1 + col0] = v10; C[r1 + col0 + 16] = v11;
        } else {
            _Float16* C = (_Float16*)Cv + cOff;
            C[r0 + col0] = (_Float16)v00; C[r0 + col0 + 16] = (_Float16)v01;
            C[r1 + col0] = (_Float16)v10; C[r1 + col0 + 16] = (_Float16)v11;
        }
    }
}

// ---------------------------------------------------------------------------
// f32 -> f16 weight conversion (vectorized, 8 elems/thread)
// ---------------------------------------------------------------------------
__global__ void cvt_f32_f16_kernel(const float* __restrict__ src,
                                   _Float16* __restrict__ dst, long long n)
{
    long long i = ((long long)blockIdx.x * blockDim.x + threadIdx.x) * 8;
    if (i + 8 <= n) {
        float4 f0 = *(const float4*)(src + i);
        float4 f1 = *(const float4*)(src + i + 4);
        _Float16* d = dst + i;
        d[0] = (_Float16)f0.x; d[1] = (_Float16)f0.y;
        d[2] = (_Float16)f0.z; d[3] = (_Float16)f0.w;
        d[4] = (_Float16)f1.x; d[5] = (_Float16)f1.y;
        d[6] = (_Float16)f1.z; d[7] = (_Float16)f1.w;
    } else {
        for (; i < n; ++i) dst[i] = (_Float16)src[i];
    }
}

// ---------------------------------------------------------------------------
// Embedding + sinusoidal positional encoding (f16 out)
// ---------------------------------------------------------------------------
__global__ void embed_kernel(const int* __restrict__ tok,
                             const float* __restrict__ emb,
                             _Float16* __restrict__ out)
{
    const int   row  = blockIdx.x;          // 0..ROWS-1
    const int   pos  = row & (SEQ - 1);
    const int   tk   = tok[row];
    const float sq   = 22.62741699796952f;  // sqrt(512)
    for (int d = threadIdx.x; d < DMODEL; d += blockDim.x) {
        float e   = emb[(long long)tk * DMODEL + d] * sq;
        float di  = (float)(d & ~1);
        float ang = (float)pos * __powf(10000.0f, -di / (float)DMODEL);
        float pe  = (d & 1) ? __cosf(ang) : __sinf(ang);
        out[(long long)row * DMODEL + d] = (_Float16)(e + pe);
    }
}

// ---------------------------------------------------------------------------
// LayerNorm over last dim (512) with fused residual; f16 in/out, f32 math
// ---------------------------------------------------------------------------
__global__ __launch_bounds__(256) void ln_kernel(
    const _Float16* __restrict__ x, const _Float16* __restrict__ res,
    const float* __restrict__ g, const float* __restrict__ b,
    _Float16* __restrict__ out)
{
    __shared__ float s1[256], s2[256];
    const int row = blockIdx.x;
    const _Float16* xr = x + (long long)row * DMODEL;
    const _Float16* rr = res ? res + (long long)row * DMODEL : nullptr;
    float v0[2], sum = 0.f, sq = 0.f;
    #pragma unroll
    for (int i = 0; i < 2; ++i) {
        int d = threadIdx.x + i * 256;
        float v = (float)xr[d] + (rr ? (float)rr[d] : 0.f);
        v0[i] = v; sum += v; sq += v * v;
    }
    s1[threadIdx.x] = sum; s2[threadIdx.x] = sq;
    __syncthreads();
    for (int off = 128; off > 0; off >>= 1) {
        if (threadIdx.x < off) {
            s1[threadIdx.x] += s1[threadIdx.x + off];
            s2[threadIdx.x] += s2[threadIdx.x + off];
        }
        __syncthreads();
    }
    const float mean = s1[0] * (1.0f / DMODEL);
    const float var  = s2[0] * (1.0f / DMODEL) - mean * mean;
    const float rstd = rsqrtf(var + 1e-5f);
    #pragma unroll
    for (int i = 0; i < 2; ++i) {
        int d = threadIdx.x + i * 256;
        out[(long long)row * DMODEL + d] = (_Float16)((v0[i] - mean) * rstd * g[d] + b[d]);
    }
}

// ---------------------------------------------------------------------------
// Masked softmax over k (S=1024). scores layout [B,H,S,S] (f16), in place.
// Faithful convention: s += mask * (-1e9); lookahead: mask = max(tril, pad).
// ---------------------------------------------------------------------------
__global__ __launch_bounds__(256) void softmax_mask_kernel(
    _Float16* __restrict__ scores, const int* __restrict__ tok, int lookahead)
{
    __shared__ float red[256];
    const long long rowid = blockIdx.x;          // (b*H + h)*S + q
    const int q  = (int)(rowid & (SEQ - 1));
    const int bh = (int)(rowid >> 10);
    const int b  = bh / NHEAD;
    _Float16* row = scores + rowid * SEQ;
    const int* tb = tok + (long long)b * SEQ;

    float v[4], mx = -1e30f;
    #pragma unroll
    for (int i = 0; i < 4; ++i) {
        int k = threadIdx.x + i * 256;
        float m = (tb[k] != 0) ? 1.f : 0.f;
        if (lookahead && k <= q) m = 1.f;
        float s = (float)row[k] + m * (-1.0e9f);
        v[i] = s; mx = fmaxf(mx, s);
    }
    red[threadIdx.x] = mx;
    __syncthreads();
    for (int off = 128; off > 0; off >>= 1) {
        if (threadIdx.x < off) red[threadIdx.x] = fmaxf(red[threadIdx.x], red[threadIdx.x + off]);
        __syncthreads();
    }
    mx = red[0];
    __syncthreads();
    float sum = 0.f;
    #pragma unroll
    for (int i = 0; i < 4; ++i) { v[i] = __expf(v[i] - mx); sum += v[i]; }
    red[threadIdx.x] = sum;
    __syncthreads();
    for (int off = 128; off > 0; off >>= 1) {
        if (threadIdx.x < off) red[threadIdx.x] += red[threadIdx.x + off];
        __syncthreads();
    }
    const float inv = 1.0f / red[0];
    #pragma unroll
    for (int i = 0; i < 4; ++i) {
        int k = threadIdx.x + i * 256;
        row[k] = (_Float16)(v[i] * inv);
    }
}

// ---------------------------------------------------------------------------
// Host-side orchestration
// ---------------------------------------------------------------------------
static void gemm(hipStream_t s, const _Float16* A, const _Float16* B, void* C,
                 int M, int N, int K, int lda, int ldb, int ldc,
                 long long sAb, long long sAh, long long sBb, long long sBh,
                 long long sCb, long long sCh, int nh, int nz,
                 float alpha, const float* bias, int flags)
{
    dim3 g(N / 64, M / 128, nz), blk(256);
    wmma_gemm_kernel<<<g, blk, 0, s>>>(A, B, C, M, N, K, lda, ldb, ldc,
                                       sAb, sAh, sBb, sBh, sCb, sCh, nh,
                                       alpha, bias, flags);
}

static void cvt(hipStream_t s, const float* src, _Float16* dst, long long n)
{
    long long chunks = (n + 7) / 8;
    int blocks = (int)((chunks + 255) / 256);
    cvt_f32_f16_kernel<<<blocks, 256, 0, s>>>(src, dst, n);
}

// Input index map — JAX pytree flatten order (dict keys sorted alphabetically)
enum {
    IN_ENC_X = 0, IN_DEC_X, IN_EMB,
    E_FFN_B1, E_FFN_B2, E_FFN_W1, E_FFN_W2,
    E_LN1_B, E_LN1_G, E_LN2_B, E_LN2_G,
    E_MHA_WK_B, E_MHA_WK_W, E_MHA_WO_B, E_MHA_WO_W,
    E_MHA_WQ_B, E_MHA_WQ_W, E_MHA_WV_B, E_MHA_WV_W,
    D_FFN_B1, D_FFN_B2, D_FFN_W1, D_FFN_W2,
    D_LN1_B, D_LN1_G, D_LN2_B, D_LN2_G, D_LN3_B, D_LN3_G,
    D_M1_WK_B, D_M1_WK_W, D_M1_WO_B, D_M1_WO_W,
    D_M1_WQ_B, D_M1_WQ_W, D_M1_WV_B, D_M1_WV_W,
    D_M2_WK_B, D_M2_WK_W, D_M2_WO_B, D_M2_WO_W,
    D_M2_WQ_B, D_M2_WQ_W, D_M2_WV_B, D_M2_WV_W,
    IN_DENSE_W, IN_DENSE_B
};

struct MW { const float *qw, *qb, *kw, *kb, *vw, *vb, *ow, *ob; };

static MW slice_mha(void* const* d_in, int base, int layer)
{
    auto F = [&](int j) { return (const float*)d_in[j]; };
    MW w;
    w.kb = F(base + 0) + (size_t)layer * DMODEL;
    w.kw = F(base + 1) + (size_t)layer * DMODEL * DMODEL;
    w.ob = F(base + 2) + (size_t)layer * DMODEL;
    w.ow = F(base + 3) + (size_t)layer * DMODEL * DMODEL;
    w.qb = F(base + 4) + (size_t)layer * DMODEL;
    w.qw = F(base + 5) + (size_t)layer * DMODEL * DMODEL;
    w.vb = F(base + 6) + (size_t)layer * DMODEL;
    w.vw = F(base + 7) + (size_t)layer * DMODEL * DMODEL;
    return w;
}

static void run_mha(hipStream_t s, const MW& w,
                    const _Float16* qin, const _Float16* kin, const _Float16* vin,
                    const int* masktok, int lookahead, _Float16* WB,
                    _Float16* Q, _Float16* Kb, _Float16* Vb,
                    _Float16* SC, _Float16* CTX, _Float16* OUT)
{
    const long long SD = (long long)SEQ * DMODEL;
    const long long SS = (long long)SEQ * SEQ;
    const long long DD = (long long)DMODEL * DMODEL;
    // projections [2048,512] @ [512,512]  (weights converted to f16 in WB)
    cvt(s, w.qw, WB, DD);
    gemm(s, qin, WB, Q,  ROWS, DMODEL, DMODEL, DMODEL, DMODEL, DMODEL,
         0,0,0,0,0,0, 1, 1, 1.f, w.qb, 0);
    cvt(s, w.kw, WB, DD);
    gemm(s, kin, WB, Kb, ROWS, DMODEL, DMODEL, DMODEL, DMODEL, DMODEL,
         0,0,0,0,0,0, 1, 1, 1.f, w.kb, 0);
    cvt(s, w.vw, WB, DD);
    gemm(s, vin, WB, Vb, ROWS, DMODEL, DMODEL, DMODEL, DMODEL, DMODEL,
         0,0,0,0,0,0, 1, 1, 1.f, w.vb, 0);
    // scores[b,h] = Q[b,h] @ K[b,h]^T / sqrt(depth)   (batched, B transposed)
    gemm(s, Q, Kb, SC, SEQ, SEQ, DEPTH, DMODEL, DMODEL, SEQ,
         SD, DEPTH, SD, DEPTH, (long long)NHEAD * SS, SS,
         NHEAD, BATCH * NHEAD, 0.125f, nullptr, GF_BT);
    softmax_mask_kernel<<<BATCH * NHEAD * SEQ, 256, 0, s>>>(SC, masktok, lookahead);
    // ctx[b,h] = attn @ V[b,h], written directly into [B,S,D] layout
    gemm(s, SC, Vb, CTX, SEQ, DEPTH, SEQ, SEQ, DMODEL, DMODEL,
         (long long)NHEAD * SS, SS, SD, DEPTH, SD, DEPTH,
         NHEAD, BATCH * NHEAD, 1.f, nullptr, 0);
    // output projection
    cvt(s, w.ow, WB, DD);
    gemm(s, CTX, WB, OUT, ROWS, DMODEL, DMODEL, DMODEL, DMODEL, DMODEL,
         0,0,0,0,0,0, 1, 1, 1.f, w.ob, 0);
}

extern "C" void kernel_launch(void* const* d_in, const int* in_sizes, int n_in,
                              void* d_out, int out_size, void* d_ws, size_t ws_size,
                              hipStream_t stream)
{
    (void)in_sizes; (void)n_in; (void)out_size; (void)ws_size;
    auto F = [&](int j) { return (const float*)d_in[j]; };
    const int* enc_x = (const int*)d_in[IN_ENC_X];
    const int* dec_x = (const int*)d_in[IN_DEC_X];
    const float* emb = F(IN_EMB);

    // ---- workspace carve (f16 elements) ----
    _Float16* p   = (_Float16*)d_ws;
    _Float16* X   = p; p += (size_t)ROWS * DMODEL;   // encoder acts / enc_out
    _Float16* XD  = p; p += (size_t)ROWS * DMODEL;   // decoder acts
    _Float16* Q   = p; p += (size_t)ROWS * DMODEL;
    _Float16* Kb  = p; p += (size_t)ROWS * DMODEL;
    _Float16* Vb  = p; p += (size_t)ROWS * DMODEL;
    _Float16* CTX = p; p += (size_t)ROWS * DMODEL;
    _Float16* MHA = p; p += (size_t)ROWS * DMODEL;
    _Float16* A1  = p; p += (size_t)ROWS * DMODEL;
    _Float16* A2  = p; p += (size_t)ROWS * DMODEL;
    _Float16* FFO = p; p += (size_t)ROWS * DMODEL;
    _Float16* MID = p; p += (size_t)ROWS * FFDIM;                 // 8 MB
    _Float16* SC  = p; p += (size_t)BATCH * NHEAD * SEQ * SEQ;    // 32 MB
    _Float16* WB  = p; p += (size_t)DMODEL * VOCAB;               // 32 MB (max B)

    const long long DF = (long long)DMODEL * FFDIM;

    // ================= Encoder =================
    embed_kernel<<<ROWS, 256, 0, stream>>>(enc_x, emb, X);
    for (int i = 0; i < NLAYER; ++i) {
        MW w = slice_mha(d_in, E_MHA_WK_B, i);
        run_mha(stream, w, X, X, X, enc_x, 0, WB, Q, Kb, Vb, SC, CTX, MHA);
        ln_kernel<<<ROWS, 256, 0, stream>>>(X, MHA,
            F(E_LN1_G) + (size_t)i * DMODEL, F(E_LN1_B) + (size_t)i * DMODEL, X);
        cvt(stream, F(E_FFN_W1) + (size_t)i * DF, WB, DF);
        gemm(stream, X, WB, MID, ROWS, FFDIM, DMODEL, DMODEL, FFDIM, FFDIM,
             0,0,0,0,0,0, 1, 1, 1.f, F(E_FFN_B1) + (size_t)i * FFDIM, GF_RELU);
        cvt(stream, F(E_FFN_W2) + (size_t)i * DF, WB, DF);
        gemm(stream, MID, WB, FFO, ROWS, DMODEL, FFDIM, FFDIM, DMODEL, DMODEL,
             0,0,0,0,0,0, 1, 1, 1.f, F(E_FFN_B2) + (size_t)i * DMODEL, 0);
        ln_kernel<<<ROWS, 256, 0, stream>>>(X, FFO,
            F(E_LN2_G) + (size_t)i * DMODEL, F(E_LN2_B) + (size_t)i * DMODEL, X);
    }
    // X now holds enc_out

    // ================= Decoder =================
    embed_kernel<<<ROWS, 256, 0, stream>>>(dec_x, emb, XD);
    for (int i = 0; i < NLAYER; ++i) {
        MW w1 = slice_mha(d_in, D_M1_WK_B, i);
        MW w2 = slice_mha(d_in, D_M2_WK_B, i);
        // masked self-attention
        run_mha(stream, w1, XD, XD, XD, dec_x, 1, WB, Q, Kb, Vb, SC, CTX, MHA);
        ln_kernel<<<ROWS, 256, 0, stream>>>(XD, MHA,
            F(D_LN1_G) + (size_t)i * DMODEL, F(D_LN1_B) + (size_t)i * DMODEL, A1);
        // cross-attention (faithful: q_in=enc_out, k_in=enc_out, v_in=a1, mask=dec_pad)
        run_mha(stream, w2, X, X, A1, dec_x, 0, WB, Q, Kb, Vb, SC, CTX, MHA);
        ln_kernel<<<ROWS, 256, 0, stream>>>(A1, MHA,
            F(D_LN2_G) + (size_t)i * DMODEL, F(D_LN2_B) + (size_t)i * DMODEL, A2);
        // FFN
        cvt(stream, F(D_FFN_W1) + (size_t)i * DF, WB, DF);
        gemm(stream, A2, WB, MID, ROWS, FFDIM, DMODEL, DMODEL, FFDIM, FFDIM,
             0,0,0,0,0,0, 1, 1, 1.f, F(D_FFN_B1) + (size_t)i * FFDIM, GF_RELU);
        cvt(stream, F(D_FFN_W2) + (size_t)i * DF, WB, DF);
        gemm(stream, MID, WB, FFO, ROWS, DMODEL, FFDIM, FFDIM, DMODEL, DMODEL,
             0,0,0,0,0,0, 1, 1, 1.f, F(D_FFN_B2) + (size_t)i * DMODEL, 0);
        ln_kernel<<<ROWS, 256, 0, stream>>>(A2, FFO,
            F(D_LN3_G) + (size_t)i * DMODEL, F(D_LN3_B) + (size_t)i * DMODEL, XD);
    }

    // ================= Final projection to vocab (f32 out) =================
    cvt(stream, F(IN_DENSE_W), WB, (long long)DMODEL * VOCAB);
    gemm(stream, XD, WB, (float*)d_out,
         ROWS, VOCAB, DMODEL, DMODEL, VOCAB, VOCAB,
         0,0,0,0,0,0, 1, 1, 1.f, F(IN_DENSE_B), GF_OUTF32);
}
